// Model_84404697301440
// MI455X (gfx1250) — matrix operand order repaired and checked by reading, not compiled, not run on hardware
//
#include <hip/hip_runtime.h>
#include <hip/hip_bf16.h>

// ---------------------------------------------------------------------------
// Autoformer forward for MI455X (gfx1250).
// GEMMs: v_wmma_f32_16x16x32_bf16, f32 accumulate. Weights are converted to
// bf16 (pre-transposed [N][K]) once per launch; activations converted once per
// GEMM. The GEMM hot loop is pure b128 load / b128 LDS store / ds_load_2addr /
// WMMA -- no VALU conversion work competing with the matrix pipe.
// Autocorrelation uses the algebraic collapse: only mean-over-(H,E) of the
// per-channel circular correlation is needed -> direct time-domain kernel.
// ---------------------------------------------------------------------------

typedef __bf16 bf16_t;
typedef bf16_t v16bf __attribute__((ext_vector_type(16)));
typedef float  v8f   __attribute__((ext_vector_type(8)));

#define BM 128
#define BN 64
#define BKK 32
#define LDSTR 40   // bf16 stride: 80 B/row -> 16B-aligned b128 staging stores

__device__ __forceinline__ unsigned short f32_to_bf16(float f) {
  unsigned u = __float_as_uint(f);
  unsigned r = (u + 0x7FFFu + ((u >> 16) & 1u)) >> 16;   // round-nearest-even
  return (unsigned short)r;
}

union FragBF { v16bf v; unsigned u[8]; };
union AccF   { v8f   v; float f[8]; };

// out = act( A(MxK,bf16) * Wt(NxK,bf16)^T + bias [+ resid] )
// M % BM == 0, N % BN == 0, K % BKK == 0 at every call site.
template <bool HAS_RES, bool HAS_GELU>
__global__ __launch_bounds__(256) void wmma_gemm_kernel(
    const unsigned short* __restrict__ A, const unsigned short* __restrict__ Wt,
    const float* __restrict__ bias, const float* __restrict__ resid,
    float* __restrict__ out, int M, int N, int K)
{
  __shared__ unsigned short lA[BM * LDSTR];
  __shared__ unsigned short lB[BN * LDSTR];
  const int tid  = threadIdx.x;
  const int n0   = blockIdx.x * BN;
  const int m0   = blockIdx.y * BM;
  const int wave = tid >> 5, lane = tid & 31;
  const int waveM = wave & 3, waveN = wave >> 2;      // 4x2 wave grid
  const int half = lane >> 4, l16 = lane & 15;

  // staging coordinates: 8 bf16 (16B) per segment
  const int sRow = tid >> 2;            // 0..63
  const int sK8  = (tid & 3) << 3;      // 0,8,16,24

  AccF acc[2][2];
#pragma unroll
  for (int i = 0; i < 2; ++i)
#pragma unroll
    for (int j = 0; j < 2; ++j)
#pragma unroll
      for (int v = 0; v < 8; ++v) acc[i][j].f[v] = 0.f;

  uint4 aReg0, aReg1, bReg;
  // prologue: fetch tile k0 = 0 into registers
  aReg0 = *(const uint4*)(A  + (size_t)(m0 + sRow)      * K + sK8);
  aReg1 = *(const uint4*)(A  + (size_t)(m0 + sRow + 64) * K + sK8);
  bReg  = *(const uint4*)(Wt + (size_t)(n0 + sRow)      * K + sK8);

  const int ksteps = K / BKK;
  for (int ks = 0; ks < ksteps; ++ks) {
    // store staged registers into LDS (b128)
    *(uint4*)&lA[(sRow)      * LDSTR + sK8] = aReg0;
    *(uint4*)&lA[(sRow + 64) * LDSTR + sK8] = aReg1;
    *(uint4*)&lB[(sRow)      * LDSTR + sK8] = bReg;
    __syncthreads();

    // issue next tile's global loads early (overlap with WMMA below)
    if (ks + 1 < ksteps) {
      int k0 = (ks + 1) * BKK;
      aReg0 = *(const uint4*)(A  + (size_t)(m0 + sRow)      * K + k0 + sK8);
      aReg1 = *(const uint4*)(A  + (size_t)(m0 + sRow + 64) * K + k0 + sK8);
      bReg  = *(const uint4*)(Wt + (size_t)(n0 + sRow)      * K + k0 + sK8);
    }

    FragBF af[2], bf[2];
#pragma unroll
    for (int s = 0; s < 2; ++s) {
      int row = waveM * 32 + s * 16 + l16;
#pragma unroll
      for (int v = 0; v < 8; ++v) {
        int klo = (v < 4 ? 2 * v : 16 + 2 * (v - 4)) + half * 8;
        af[s].u[v] = *(const unsigned*)&lA[row * LDSTR + klo];
      }
      int n = waveN * 32 + s * 16 + l16;
#pragma unroll
      for (int v = 0; v < 8; ++v) {
        int klo = (v < 4 ? 2 * v : 16 + 2 * (v - 4)) + half * 8;
        bf[s].u[v] = *(const unsigned*)&lB[n * LDSTR + klo];
      }
    }
#pragma unroll
    for (int i = 0; i < 2; ++i)
#pragma unroll
      for (int j = 0; j < 2; ++j)
        acc[i][j].v = __builtin_amdgcn_wmma_f32_16x16x32_bf16(
            false, af[i].v, false, bf[j].v, (short)0, acc[i][j].v, false, false);
    __syncthreads();
  }

  // epilogue: bias (+ residual) (+ GELU) -- branch-free, fully covered tiles
#pragma unroll
  for (int j = 0; j < 2; ++j) {
    int n = n0 + waveN * 32 + j * 16 + l16;
    float bn = bias[n];
#pragma unroll
    for (int i = 0; i < 2; ++i) {
#pragma unroll
      for (int v = 0; v < 8; ++v) {
        int m = m0 + waveM * 32 + i * 16 + v + half * 8;
        size_t o = (size_t)m * N + n;
        float val = acc[i][j].f[v] + bn;
        if (HAS_RES) val += resid[o];
        if (HAS_GELU) val = 0.5f * val * (1.f + erff(val * 0.70710678118f));
        out[o] = val;
      }
    }
  }
}

// weight f32 (KxN, row-major) -> bf16 transposed (NxK)
__global__ void convert_w_kernel(const float* __restrict__ src,
                                 unsigned short* __restrict__ dst, int K, int N)
{
  int idx = blockIdx.x * blockDim.x + threadIdx.x;
  if (idx >= K * N) return;
  int k = idx / N, n = idx % N;                 // coalesced read
  dst[(size_t)n * K + k] = f32_to_bf16(src[idx]);
}

// activation f32 -> bf16 flat (packed pair per thread)
__global__ void convert_a_kernel(const float2* __restrict__ src,
                                 unsigned* __restrict__ dst, int npairs)
{
  int idx = blockIdx.x * blockDim.x + threadIdx.x;
  if (idx >= npairs) return;
  float2 v = src[idx];
  dst[idx] = (unsigned)f32_to_bf16(v.x) | ((unsigned)f32_to_bf16(v.y) << 16);
}

// series_decomp: 25-tap edge-replicated moving average. mode: 0=seasonal only,
// 1=write trend, 2=accumulate trend
__global__ void decomp_kernel(const float* __restrict__ x,
                              float* __restrict__ seas, float* __restrict__ trend,
                              int total, int L, int C, int mode)
{
  int idx = blockIdx.x * blockDim.x + threadIdx.x;
  if (idx >= total) return;
  int c = idx % C;
  int l = (idx / C) % L;
  int b = idx / (C * L);
  const float* xb = x + (size_t)b * L * C;
  float s = 0.f;
#pragma unroll 5
  for (int j = -12; j <= 12; ++j) {
    int t = l + j;
    t = t < 0 ? 0 : (t > L - 1 ? L - 1 : t);
    s += xb[(size_t)t * C + c];
  }
  float m = s * (1.f / 25.f);
  seas[idx] = x[idx] - m;
  if (mode == 1) trend[idx] = m;
  else if (mode == 2) trend[idx] += m;
}

// meanCorr[b,tau] = (1/512) * sum_t dot(Q[b,(t+tau)%512,:], K[b,t,:])
__global__ __launch_bounds__(256) void mean_corr_kernel(
    const float* __restrict__ q, const float* __restrict__ k,
    float* __restrict__ corr)
{
  __shared__ float lk[512];
  __shared__ float red[256];
  int b = blockIdx.x;
  int tau = blockIdx.y * 16 + (threadIdx.x >> 4);
  int c = threadIdx.x & 15;
  int d0 = c * 32;
  const float* qb = q + (size_t)b * 512 * 512;
  const float* kb = k + (size_t)b * 512 * 512;
  float acc = 0.f;
  for (int t = 0; t < 512; ++t) {
    __syncthreads();
    lk[threadIdx.x]       = kb[(size_t)t * 512 + threadIdx.x];
    lk[threadIdx.x + 256] = kb[(size_t)t * 512 + threadIdx.x + 256];
    __syncthreads();
    const float* qr = qb + (size_t)((t + tau) & 511) * 512 + d0;
#pragma unroll 8
    for (int j = 0; j < 32; ++j) acc += qr[j] * lk[d0 + j];
  }
  red[threadIdx.x] = acc;
  __syncthreads();
  for (int off = 8; off > 0; off >>= 1) {
    if (c < off) red[threadIdx.x] += red[threadIdx.x + off];
    __syncthreads();
  }
  if (c == 0) corr[b * 512 + tau] = red[threadIdx.x] * (1.f / 512.f);
}

// top-18 + softmax per batch
__global__ void topk_kernel(const float* __restrict__ corr,
                            int* __restrict__ delay, float* __restrict__ wout)
{
  __shared__ float v[512];
  __shared__ float wv[18];
  __shared__ int   wd[18];
  int b = blockIdx.x, tid = threadIdx.x;
  v[tid]       = corr[b * 512 + tid];
  v[tid + 256] = corr[b * 512 + tid + 256];
  __syncthreads();
  if (tid == 0) {
    for (int i = 0; i < 18; ++i) {
      float best = -3.4e38f; int bi = 0;
      for (int t = 0; t < 512; ++t)
        if (v[t] > best) { best = v[t]; bi = t; }
      wv[i] = best; wd[i] = bi; v[bi] = -3.4e38f;
    }
    float mx = wv[0], s = 0.f;
    for (int i = 0; i < 18; ++i) { wv[i] = expf(wv[i] - mx); s += wv[i]; }
    float inv = 1.f / s;
    for (int i = 0; i < 18; ++i) {
      wout[b * 18 + i]  = wv[i] * inv;
      delay[b * 18 + i] = wd[i];
    }
  }
}

// agg[b,t,:] = sum_i w[b,i] * V[b,(t+delay[b,i])%512,:]
__global__ __launch_bounds__(256) void agg_kernel(
    const float* __restrict__ v, const int* __restrict__ delay,
    const float* __restrict__ w, float* __restrict__ out)
{
  __shared__ float lw[18];
  __shared__ int   ld[18];
  int b = blockIdx.x >> 9, t = blockIdx.x & 511;
  int tid = threadIdx.x;
  if (tid < 18) { lw[tid] = w[b * 18 + tid]; ld[tid] = delay[b * 18 + tid]; }
  __syncthreads();
  const float* vb = v + (size_t)b * 512 * 512;
#pragma unroll
  for (int h = 0; h < 2; ++h) {
    int d = tid + h * 256;
    float acc = 0.f;
    for (int i = 0; i < 18; ++i)
      acc += lw[i] * vb[(size_t)((t + ld[i]) & 511) * 512 + d];
    out[((size_t)b * 512 + t) * 512 + d] = acc;
  }
}

// token circular conv (C_IN=21 -> 512) + temporal embed (4 -> 512)
__global__ __launch_bounds__(256) void embed_kernel(
    const float* __restrict__ x, const float* __restrict__ mark,
    const float* __restrict__ tok, const float* __restrict__ temp,
    float* __restrict__ out)
{
  __shared__ float lx[63];
  __shared__ float lm[4];
  int b = blockIdx.x >> 9, l = blockIdx.x & 511;
  int tid = threadIdx.x;
  if (tid < 63) {
    int kk = tid / 21, i = tid % 21;
    int row = (l + kk - 1 + 512) & 511;
    lx[tid] = x[((size_t)b * 512 + row) * 21 + i];
  }
  if (tid >= 64 && tid < 68)
    lm[tid - 64] = mark[((size_t)b * 512 + l) * 4 + (tid - 64)];
  __syncthreads();
#pragma unroll
  for (int h = 0; h < 2; ++h) {
    int o = tid + h * 256;
    const float* wt = tok + (size_t)o * 63;   // tok[o][i][k], 21*3
    float acc = 0.f;
    for (int i = 0; i < 21; ++i)
#pragma unroll
      for (int kk = 0; kk < 3; ++kk)
        acc += lx[kk * 21 + i] * wt[i * 3 + kk];
#pragma unroll
    for (int f = 0; f < 4; ++f) acc += lm[f] * temp[f * 512 + o];
    out[((size_t)b * 512 + l) * 512 + o] = acc;
  }
}

// layernorm stage 1: per-row LN over D=512
__global__ __launch_bounds__(256) void ln_row_kernel(
    const float* __restrict__ x, const float* __restrict__ g,
    const float* __restrict__ bta, float* __restrict__ out)
{
  __shared__ float red[256];
  int row = blockIdx.x, tid = threadIdx.x;
  const float* xr = x + (size_t)row * 512;
  float a = xr[tid], b2 = xr[tid + 256];
  red[tid] = a + b2;
  __syncthreads();
  for (int off = 128; off; off >>= 1) {
    if (tid < off) red[tid] += red[tid + off];
    __syncthreads();
  }
  float mu = red[0] * (1.f / 512.f);
  __syncthreads();
  float da = a - mu, db = b2 - mu;
  red[tid] = da * da + db * db;
  __syncthreads();
  for (int off = 128; off; off >>= 1) {
    if (tid < off) red[tid] += red[tid + off];
    __syncthreads();
  }
  float inv = rsqrtf(red[0] * (1.f / 512.f) + 1e-5f);
  out[(size_t)row * 512 + tid]       = da * inv * g[tid] + bta[tid];
  out[(size_t)row * 512 + tid + 256] = db * inv * g[tid + 256] + bta[tid + 256];
}

// layernorm stage 2: subtract per-(b,d) mean over L
__global__ __launch_bounds__(256) void ln_colmean_kernel(
    const float* __restrict__ x, float* __restrict__ out)
{
  int b = blockIdx.x;
  int d = blockIdx.y * 256 + threadIdx.x;
  const float* xb = x + (size_t)b * 512 * 512;
  float s = 0.f;
  for (int l = 0; l < 512; ++l) s += xb[(size_t)l * 512 + d];
  float mu = s * (1.f / 512.f);
  float* ob = out + (size_t)b * 512 * 512;
  for (int l = 0; l < 512; ++l)
    ob[(size_t)l * 512 + d] = xb[(size_t)l * 512 + d] - mu;
}

// trend += circ_conv1d(trend_sum(B,512,512), proj_w(21,512,3))
__global__ __launch_bounds__(256) void trendproj_kernel(
    const float* __restrict__ ts, const float* __restrict__ pw,
    float* __restrict__ trend)
{
  __shared__ float rows[3 * 512];
  int b = blockIdx.x >> 9, l = blockIdx.x & 511;
  int tid = threadIdx.x;
#pragma unroll
  for (int r = 0; r < 6; ++r) {
    int e = tid + r * 256;
    int kk = e >> 9, i = e & 511;
    int row = (l + kk - 1 + 512) & 511;
    rows[e] = ts[((size_t)b * 512 + row) * 512 + i];
  }
  __syncthreads();
  int wave = tid >> 5, lane = tid & 31;
  for (int o = wave; o < 21; o += 8) {
    const float* w = pw + (size_t)o * 1536;   // pw[o][i][k]
    float acc = 0.f;
    for (int i = lane; i < 512; i += 32) {
      acc += rows[i]        * w[i * 3 + 0];
      acc += rows[512 + i]  * w[i * 3 + 1];
      acc += rows[1024 + i] * w[i * 3 + 2];
    }
    for (int off = 16; off; off >>= 1) acc += __shfl_xor(acc, off, 32);
    if (lane == 0) trend[((size_t)b * 512 + l) * 21 + o] += acc;
  }
}

__global__ void mean21_kernel(const float* __restrict__ x, float* __restrict__ m)
{
  int b = blockIdx.x, c = threadIdx.x;
  if (c >= 21) return;
  float s = 0.f;
  for (int l = 0; l < 512; ++l) s += x[((size_t)b * 512 + l) * 21 + c];
  m[b * 21 + c] = s * (1.f / 512.f);
}

// seasonal_init = [seasonal[:, -256:], 0]; trend_init = [trend[:, -256:], mean]
__global__ void build_init_kernel(
    const float* __restrict__ seas, const float* __restrict__ tr,
    const float* __restrict__ mean, float* __restrict__ sinit,
    float* __restrict__ tinit)
{
  int idx = blockIdx.x * blockDim.x + threadIdx.x;
  if (idx >= 16 * 512 * 21) return;
  int c = idx % 21;
  int l = (idx / 21) % 512;
  int b = idx / (21 * 512);
  if (l < 256) {
    size_t s = ((size_t)b * 512 + 256 + l) * 21 + c;
    sinit[idx] = seas[s];
    tinit[idx] = tr[s];
  } else {
    sinit[idx] = 0.f;
    tinit[idx] = mean[b * 21 + c];
  }
}

// dec_out(8192x512) @ dec_proj_w(512x21) + b
__global__ void proj21_kernel(const float* __restrict__ x,
                              const float* __restrict__ w,
                              const float* __restrict__ bias,
                              float* __restrict__ out)
{
  int idx = blockIdx.x * blockDim.x + threadIdx.x;
  if (idx >= 8192 * 21) return;
  int o = idx % 21, m = idx / 21;
  const float* xr = x + (size_t)m * 512;
  float acc = bias[o];
  for (int d = 0; d < 512; ++d) acc += xr[d] * w[d * 21 + o];
  out[idx] = acc;
}

__global__ void final_kernel(const float* __restrict__ trend,
                             const float* __restrict__ sp,
                             float* __restrict__ out)
{
  int idx = blockIdx.x * blockDim.x + threadIdx.x;
  if (idx >= 16 * 256 * 21) return;
  int c = idx % 21;
  int t = (idx / 21) % 256;
  int b = idx / (21 * 256);
  size_t s = ((size_t)b * 512 + 256 + t) * 21 + c;
  out[idx] = trend[s] + sp[s];
}

// ---------------------------------------------------------------------------
extern "C" void kernel_launch(void* const* d_in, const int* in_sizes, int n_in,
                              void* d_out, int out_size, void* d_ws, size_t ws_size,
                              hipStream_t stream)
{
  if (n_in < 59) return;
  auto P = [&](int i) { return (const float*)d_in[i]; };

  const int M = 16 * 512;               // flattened B*L rows
  const size_t NBIG = (size_t)M * 512;  // 4,194,304 floats

  float* ws = (float*)d_ws;
  float* X    = ws + 0 * NBIG;   // current activations
  float* Q    = ws + 1 * NBIG;   // q / agg scratch
  float* Kb   = ws + 2 * NBIG;
  float* Vb   = ws + 3 * NBIG;
  float* TMP  = ws + 4 * NBIG;   // ping-pong
  float* ENCO = ws + 5 * NBIG;   // saved encoder output
  float* TRS  = ws + 6 * NBIG;   // decoder trend sum (B,512,512)
  float* HID  = ws + 7 * NBIG;   // FFN hidden (B,512,2048) -> 4*NBIG
  size_t off  = 11 * NBIG;
  float* CORR   = ws + off; off += 16 * 512;
  int*   DELAY  = (int*)(ws + off); off += 16 * 18 + 8;
  float* WEI    = ws + off; off += 16 * 18 + 8;
  float* SEAS21 = ws + off; off += 16 * 512 * 21 + 8;
  float* TR21   = ws + off; off += 16 * 512 * 21 + 8;
  float* MEAN21 = ws + off; off += 16 * 32;
  float* SINIT  = ws + off; off += 16 * 512 * 21 + 8;
  float* TREND  = ws + off; off += 16 * 512 * 21 + 8;
  float* SP     = ws + off; off += 16 * 512 * 21 + 8;
  // bf16 regions (ushort), 16B-aligned offsets
  off = (off + 7) & ~(size_t)7;
  unsigned short* WT   = (unsigned short*)(ws + off); off += 10485760 / 2;
  unsigned short* ABF  = (unsigned short*)(ws + off); off += ((size_t)M * 2048) / 2;
  unsigned short* ABF2 = (unsigned short*)(ws + off); off += ((size_t)M * 512) / 2;

  // ---- convert + transpose all GEMM weights to bf16 [N][K], once ----
  struct WSpec { int idx, K, N; };
  const WSpec wspec[22] = {
    {8,512,512},{10,512,512},{12,512,512},{14,512,512},{16,512,2048},{18,2048,512},
    {20,512,512},{22,512,512},{24,512,512},{26,512,512},{28,512,2048},{30,2048,512},
    {34,512,512},{36,512,512},{38,512,512},{40,512,512},
    {42,512,512},{44,512,512},{46,512,512},{48,512,512},
    {50,512,2048},{52,2048,512}};
  size_t wtoff[64];
  {
    size_t acc = 0;
    for (int i = 0; i < 22; ++i) {
      const WSpec& s = wspec[i];
      wtoff[s.idx] = acc;
      int tot = s.K * s.N;
      convert_w_kernel<<<(tot + 255) / 256, 256, 0, stream>>>(
          P(s.idx), WT + acc, s.K, s.N);
      acc += (size_t)tot;
    }
  }

  auto cvtA = [&](const float* src, unsigned short* dst, int total) {
    convert_a_kernel<<<((total / 2) + 255) / 256, 256, 0, stream>>>(
        (const float2*)src, (unsigned*)dst, total / 2);
  };
  // act/res-templated GEMM dispatch (A already bf16; weight by input index)
  auto gemmB = [&](const unsigned short* Abf, int widx, const float* bias,
                   const float* resid, float* out, int N, int K, int act) {
    dim3 g(N / BN, M / BM);
    const unsigned short* Wt = WT + wtoff[widx];
    if (resid)
      wmma_gemm_kernel<true, false><<<g, 256, 0, stream>>>(
          Abf, Wt, bias, resid, out, M, N, K);
    else if (act)
      wmma_gemm_kernel<false, true><<<g, 256, 0, stream>>>(
          Abf, Wt, bias, nullptr, out, M, N, K);
    else
      wmma_gemm_kernel<false, false><<<g, 256, 0, stream>>>(
          Abf, Wt, bias, nullptr, out, M, N, K);
  };
  auto decomp512 = [&](const float* in, float* seas, float* trend, int mode) {
    decomp_kernel<<<((int)(M * 512) + 255) / 256, 256, 0, stream>>>(
        in, seas, trend, M * 512, 512, 512, mode);
  };
  // out = xq + AutoCorrelation-attention(xq; xkv)
  auto attn = [&](const float* xq, const float* xkv, int p, float* out) {
    cvtA(xq, ABF, M * 512);
    gemmB(ABF, p + 0, P(p + 1), nullptr, Q, 512, 512, 0);
    const unsigned short* kvb = ABF;
    if (xkv != xq) { cvtA(xkv, ABF2, M * 512); kvb = ABF2; }
    gemmB(kvb, p + 2, P(p + 3), nullptr, Kb, 512, 512, 0);
    gemmB(kvb, p + 4, P(p + 5), nullptr, Vb, 512, 512, 0);
    mean_corr_kernel<<<dim3(16, 32), 256, 0, stream>>>(Q, Kb, CORR);
    topk_kernel<<<16, 256, 0, stream>>>(CORR, DELAY, WEI);
    agg_kernel<<<16 * 512, 256, 0, stream>>>(Vb, DELAY, WEI, Q);  // Q reused
    cvtA(Q, ABF, M * 512);
    gemmB(ABF, p + 6, P(p + 7), xq /*residual*/, out, 512, 512, 0);
  };

  const float* x_enc  = P(0);
  const float* xm_enc = P(1);
  const float* xm_dec = P(3);

  // ---- decomposition of raw input (C=21) + init buffers ----
  int tot21 = 16 * 512 * 21;
  decomp_kernel<<<(tot21 + 255) / 256, 256, 0, stream>>>(
      x_enc, SEAS21, TR21, tot21, 512, 21, 1);
  mean21_kernel<<<16, 32, 0, stream>>>(x_enc, MEAN21);
  build_init_kernel<<<(tot21 + 255) / 256, 256, 0, stream>>>(
      SEAS21, TR21, MEAN21, SINIT, TREND);

  // ---- encoder ----
  embed_kernel<<<16 * 512, 256, 0, stream>>>(x_enc, xm_enc, P(4), P(5), X);
  for (int layer = 0; layer < 2; ++layer) {
    int p = 8 + layer * 12;
    attn(X, X, p, TMP);                                   // TMP = X + attn(X)
    decomp512(TMP, X, nullptr, 0);                        // X = seasonal
    cvtA(X, ABF, M * 512);
    gemmB(ABF, p + 8, P(p + 9), nullptr, HID, 2048, 512, 1);       // GELU
    cvtA(HID, ABF, M * 2048);
    gemmB(ABF, p + 10, P(p + 11), X, TMP, 512, 2048, 0);           // + resid
    decomp512(TMP, X, nullptr, 0);
  }
  ln_row_kernel<<<M, 256, 0, stream>>>(X, P(32), P(33), TMP);
  ln_colmean_kernel<<<dim3(16, 2), 256, 0, stream>>>(TMP, ENCO);

  // ---- decoder ----
  embed_kernel<<<16 * 512, 256, 0, stream>>>(SINIT, xm_dec, P(6), P(7), X);
  attn(X, X, 34, TMP);                                    // self attention
  decomp512(TMP, X, TRS, 1);                              // t1 -> TRS
  attn(X, ENCO, 42, TMP);                                 // cross attention
  decomp512(TMP, X, TRS, 2);                              // t2 accumulated
  cvtA(X, ABF, M * 512);
  gemmB(ABF, 50, P(51), nullptr, HID, 2048, 512, 1);
  cvtA(HID, ABF, M * 2048);
  gemmB(ABF, 52, P(53), X, TMP, 512, 2048, 0);
  decomp512(TMP, X, TRS, 2);                              // t3 accumulated
  trendproj_kernel<<<16 * 512, 256, 0, stream>>>(TRS, P(54), TREND);

  ln_row_kernel<<<M, 256, 0, stream>>>(X, P(55), P(56), TMP);
  ln_colmean_kernel<<<dim3(16, 2), 256, 0, stream>>>(TMP, X);
  proj21_kernel<<<(8192 * 21 + 255) / 256, 256, 0, stream>>>(X, P(57), P(58), SP);
  final_kernel<<<(16 * 256 * 21 + 255) / 256, 256, 0, stream>>>(
      TREND, SP, (float*)d_out);
  (void)in_sizes; (void)out_size; (void)ws_size;
}
